// CrossTransformerBlock_79302276153474
// MI455X (gfx1250) — matrix operand order repaired and checked by reading, not compiled, not run on hardware
//
#include <hip/hip_runtime.h>
#include <hip/hip_bf16.h>

// ---------------------------------------------------------------------------
// Problem constants (from reference)
// ---------------------------------------------------------------------------
#define NPTS    8192
#define KNN     32
#define DIM     256
#define DIMH    128
#define DIMOUT  512
#define NB      2            // points per fused workgroup -> 64 rows

typedef __attribute__((ext_vector_type(16))) __bf16 bfx16;
typedef __attribute__((ext_vector_type(8)))  __bf16 bfx8;
typedef __attribute__((ext_vector_type(8)))  float  fx8;

union BF16Frag { bfx16 v; bfx8 h[2]; };

// ---------------------------------------------------------------------------
// bf16 helpers (round-to-nearest-even)
// ---------------------------------------------------------------------------
__device__ static inline unsigned short f2bfu(float f) {
    union { float f; unsigned u; } v; v.f = f;
    unsigned u = v.u + 0x7FFFu + ((v.u >> 16) & 1u);
    return (unsigned short)(u >> 16);
}
__device__ static inline __bf16 f2bf(float f) {
    unsigned short s = f2bfu(f);
    __bf16 b; __builtin_memcpy(&b, &s, 2); return b;
}
__device__ static inline float bf2f(__bf16 b) {
    unsigned short s; __builtin_memcpy(&s, &b, 2);
    union { unsigned u; float f; } v; v.u = ((unsigned)s) << 16;
    return v.f;
}

// ---------------------------------------------------------------------------
// WMMA fragment loaders (v_wmma_f32_16x16x32_bf16)
// A (16x32 bf16), row-major source, lda in elements:
//   lanes 0-15: M=lane,   K = ks*32 + {0..7, 16..23}
//   lanes16-31: M=lane-16,K = ks*32 + {8..15, 24..31}
// B (32x16 bf16) from pre-transposed weights Bt (N-major, ldb = Ktot):
//   lanes 0-15: N=lane,   K = ks*32 + {0..7, 16..23}
//   lanes16-31: N=lane-16,K = ks*32 + {8..15, 24..31}
// ---------------------------------------------------------------------------
__device__ static inline bfx16 load_afrag(const __bf16* A, int lda,
                                          int row0, int ks, int lane) {
    int m  = row0 + (lane & 15);
    int kb = ks * 32 + ((lane >> 4) & 1) * 8;
    const __bf16* p = A + m * lda + kb;
    BF16Frag u;
    u.h[0] = *(const bfx8*)(p);
    u.h[1] = *(const bfx8*)(p + 16);
    return u.v;
}
__device__ static inline bfx16 load_bfrag(const __bf16* Bt, int ldb,
                                          int col0, int ks, int lane) {
    int n  = col0 + (lane & 15);
    int kb = ks * 32 + ((lane >> 4) & 1) * 8;
    const __bf16* p = Bt + n * ldb + kb;
    BF16Frag u;
    u.h[0] = *(const bfx8*)(p);
    u.h[1] = *(const bfx8*)(p + 16);
    return u.v;
}

// ---------------------------------------------------------------------------
// Column-panel GEMM: B fragments for one 16-col tile are loaded ONCE into
// registers (NK x 8 VGPRs) and reused across all 4 row tiles (64 rows of A in
// LDS). Cuts L2 weight traffic 4x vs per-tile B loads. Epilogue is a lambda
// invoked with (row0, acc) in the documented C/D VGPR layout.
// NOTE: callers must NOT unroll their column-tile loop — exactly one B panel
// (64 VGPRs) should be live at a time to stay under 256 VGPRs (avoids
// s_set_vgpr_msb SALU overhead and keeps occupancy up).
// ---------------------------------------------------------------------------
template <int NK, typename EPI>
__device__ static inline void gemm_colpanel(const __bf16* A, int lda,
                                            const __bf16* Bt, int ldb,
                                            int col0, int lane, EPI&& epi) {
    bfx16 bfr[NK];
    #pragma unroll
    for (int ks = 0; ks < NK; ++ks)
        bfr[ks] = load_bfrag(Bt, ldb, col0, ks, lane);
    #pragma unroll
    for (int rt = 0; rt < 4; ++rt) {
        fx8 acc = {};
        #pragma unroll
        for (int ks = 0; ks < NK; ++ks) {
            bfx16 a = load_afrag(A, lda, rt * 16, ks, lane);
            acc = __builtin_amdgcn_wmma_f32_16x16x32_bf16(
                false, a, false, bfr[ks], (short)0, acc, false, false);
        }
        epi(rt * 16, acc);
    }
}

// ---------------------------------------------------------------------------
// Prep kernels: weight transpose+convert, feature convert, xyz passthrough
// ---------------------------------------------------------------------------
__global__ void k_transpose_bf16(const float* __restrict__ src,
                                 __bf16* __restrict__ dst, int IN, int OUT) {
    int total = IN * OUT;
    for (int i = blockIdx.x * blockDim.x + threadIdx.x; i < total;
         i += gridDim.x * blockDim.x) {
        int o  = i / IN;
        int ii = i - o * IN;
        dst[i] = f2bf(src[ii * OUT + o]);   // dst[o*IN+ii] = src[ii][o]
    }
}
__global__ void k_conv_bf16(const float* __restrict__ src,
                            __bf16* __restrict__ dst, int n) {
    for (int i = blockIdx.x * blockDim.x + threadIdx.x; i < n;
         i += gridDim.x * blockDim.x)
        dst[i] = f2bf(src[i]);
}
__global__ void k_copy_f32(const float* __restrict__ src,
                           float* __restrict__ dst, int n) {
    for (int i = blockIdx.x * blockDim.x + threadIdx.x; i < n;
         i += gridDim.x * blockDim.x)
        dst[i] = src[i];
}

// ---------------------------------------------------------------------------
// Standalone WMMA GEMM (Ktot = 256): C[64-row strip x Ncols] = A @ Bt + bias.
// A strip staged in LDS (32 KB) once per block; waves own column tiles with
// register-resident B panels.
// ---------------------------------------------------------------------------
__global__ __launch_bounds__(256)
void gemm_bf16(const __bf16* __restrict__ A, const __bf16* __restrict__ Bt,
               const float* __restrict__ bias, float* __restrict__ C,
               int Ncols, int ldc) {
    __shared__ __bf16 As[64 * DIM];           // 32 KB
    const int tid     = threadIdx.x;
    const int lane    = tid & 31;
    const int wave    = tid >> 5;
    const int rowBase = blockIdx.x * 64;

    // stage 64 x 256 bf16 A strip -> LDS
    {
        const uint4* src = (const uint4*)(A + (size_t)rowBase * DIM);
        uint4* dst = (uint4*)As;
        for (int i = tid; i < (64 * DIM) / 8; i += 256)
            dst[i] = src[i];
    }
    __syncthreads();

    #pragma unroll 1
    for (int ct = wave; ct * 16 < Ncols; ct += 8) {
        const int col0 = ct * 16;
        gemm_colpanel<8>(As, DIM, Bt, DIM, col0, lane,
            [&](int row0, fx8 acc) {
                const int coln = col0 + (lane & 15);
                const float bb = bias ? bias[coln] : 0.0f;
                const int rb = rowBase + row0 + ((lane >> 4) & 1) * 8;
                #pragma unroll
                for (int r = 0; r < 8; ++r)
                    C[(size_t)(rb + r) * ldc + coln] = acc[r] + bb;
            });
    }
}

// ---------------------------------------------------------------------------
// Fused attention kernel: one block handles NB=2 points (64 rows of N*K).
// LDS (dynamic, 176 KB):
//   AKV  64x256 bf16  (feats_kv tile; reused as S1)
//   R1   64x128 bf16  (relu(rel@d1+b1))
//   PE   64x256 bf16  (pos_encode; overwritten in place by VP = v+pe)
//   H    64x256 bf16  (q - k + pe)
//   SC   64x256 f32   (gamma scores, softmaxed over k)
// ---------------------------------------------------------------------------
#define LDS_AKV  0
#define LDS_R1   32768
#define LDS_PE   49152
#define LDS_H    81920
#define LDS_SC   114688
#define LDS_TOT  180224

__global__ __launch_bounds__(256)
void fused_attn(const float* __restrict__ xyz_q,
                const float* __restrict__ xyz_kv,
                const float* __restrict__ feats_kv,
                const float* __restrict__ d1w, const float* __restrict__ d1b,
                const float* __restrict__ d2b,
                const float* __restrict__ g1b, const float* __restrict__ g2b,
                const __bf16* __restrict__ WkT, const __bf16* __restrict__ WvT,
                const __bf16* __restrict__ D2T,
                const __bf16* __restrict__ G1T, const __bf16* __restrict__ G2T,
                const float* __restrict__ qattn,
                __bf16* __restrict__ resb) {
    extern __shared__ char smem[];
    __bf16* AKV = (__bf16*)(smem + LDS_AKV);
    __bf16* R1  = (__bf16*)(smem + LDS_R1);
    __bf16* PE  = (__bf16*)(smem + LDS_PE);
    __bf16* H   = (__bf16*)(smem + LDS_H);
    float*  SC  = (float*)(smem + LDS_SC);
    __bf16* S1  = AKV;   // reuse after V projection
    __bf16* VP  = PE;    // v+pe overwrites pos_encode in place

    const int tid  = threadIdx.x;
    const int lane = tid & 31;
    const int wave = tid >> 5;
    const int n0   = blockIdx.x * NB;

    // ---- Stage 0: stage feats_kv (64x256) as bf16; pos-MLP layer 1 (K=3, VALU)
    {
        const float* src = feats_kv + (size_t)(n0 * KNN) * DIM;
        for (int i = tid; i < (NB * KNN * DIM) / 4; i += 256) {
            float4 f = ((const float4*)src)[i];
            unsigned lo = (unsigned)f2bfu(f.x) | ((unsigned)f2bfu(f.y) << 16);
            unsigned hi = (unsigned)f2bfu(f.z) | ((unsigned)f2bfu(f.w) << 16);
            *(uint2*)(AKV + i * 4) = make_uint2(lo, hi);
        }
        for (int t = tid; t < NB * KNN * DIMH; t += 256) {
            int r = t >> 7, j = t & 127;
            int p = r >> 5, kk = r & 31;
            int n = n0 + p;
            float rx = xyz_q[n * 3 + 0] - xyz_kv[(n * KNN + kk) * 3 + 0];
            float ry = xyz_q[n * 3 + 1] - xyz_kv[(n * KNN + kk) * 3 + 1];
            float rz = xyz_q[n * 3 + 2] - xyz_kv[(n * KNN + kk) * 3 + 2];
            float s = rx * d1w[j] + ry * d1w[DIMH + j] + rz * d1w[2 * DIMH + j]
                      + d1b[j];
            R1[t] = f2bf(fmaxf(s, 0.0f));
        }
    }
    __syncthreads();

    // ---- Stage 1: PE = R1 @ d2T + d2_b  (K = 128)
    #pragma unroll 1
    for (int ct = wave; ct < 16; ct += 8) {
        const int col0 = ct * 16;
        gemm_colpanel<4>(R1, DIMH, D2T, DIMH, col0, lane,
            [&](int row0, fx8 acc) {
                const int coln = col0 + (lane & 15);
                const float bb = d2b[coln];
                const int rb = row0 + ((lane >> 4) & 1) * 8;
                #pragma unroll
                for (int r = 0; r < 8; ++r)
                    PE[(rb + r) * DIM + coln] = f2bf(acc[r] + bb);
            });
    }
    __syncthreads();

    // ---- Stage 2: H = q_attn - (AKV @ WkT) + PE
    #pragma unroll 1
    for (int ct = wave; ct < 16; ct += 8) {
        const int col0 = ct * 16;
        gemm_colpanel<8>(AKV, DIM, WkT, DIM, col0, lane,
            [&](int row0, fx8 acc) {
                const int coln = col0 + (lane & 15);
                const int rb = row0 + ((lane >> 4) & 1) * 8;
                #pragma unroll
                for (int r = 0; r < 8; ++r) {
                    const int rr = rb + r;
                    float qv = qattn[(size_t)(n0 + (rr >> 5)) * DIM + coln];
                    float pe = bf2f(PE[rr * DIM + coln]);
                    H[rr * DIM + coln] = f2bf(qv - acc[r] + pe);
                }
            });
    }
    __syncthreads();

    // ---- Stage 3: VP = (AKV @ WvT) + PE   (in place over PE; each element
    //               read and written only by its owning lane)
    #pragma unroll 1
    for (int ct = wave; ct < 16; ct += 8) {
        const int col0 = ct * 16;
        gemm_colpanel<8>(AKV, DIM, WvT, DIM, col0, lane,
            [&](int row0, fx8 acc) {
                const int coln = col0 + (lane & 15);
                const int rb = row0 + ((lane >> 4) & 1) * 8;
                #pragma unroll
                for (int r = 0; r < 8; ++r) {
                    const int rr = rb + r;
                    float pe = bf2f(PE[rr * DIM + coln]);
                    VP[rr * DIM + coln] = f2bf(acc[r] + pe);
                }
            });
    }
    __syncthreads();

    // ---- Stage 4: S1 = relu(H @ g1T + g1_b)   (into AKV space)
    #pragma unroll 1
    for (int ct = wave; ct < 16; ct += 8) {
        const int col0 = ct * 16;
        gemm_colpanel<8>(H, DIM, G1T, DIM, col0, lane,
            [&](int row0, fx8 acc) {
                const int coln = col0 + (lane & 15);
                const float bb = g1b[coln];
                const int rb = row0 + ((lane >> 4) & 1) * 8;
                #pragma unroll
                for (int r = 0; r < 8; ++r)
                    S1[(rb + r) * DIM + coln] = f2bf(fmaxf(acc[r] + bb, 0.0f));
            });
    }
    __syncthreads();

    // ---- Stage 5: SC = S1 @ g2T + g2_b  (fp32 scores)
    #pragma unroll 1
    for (int ct = wave; ct < 16; ct += 8) {
        const int col0 = ct * 16;
        gemm_colpanel<8>(S1, DIM, G2T, DIM, col0, lane,
            [&](int row0, fx8 acc) {
                const int coln = col0 + (lane & 15);
                const float bb = g2b[coln];
                const int rb = row0 + ((lane >> 4) & 1) * 8;
                #pragma unroll
                for (int r = 0; r < 8; ++r)
                    SC[(rb + r) * DIM + coln] = acc[r] + bb;
            });
    }
    __syncthreads();

    // ---- Stage 6: softmax over k fused with weighted reduction:
    //      res[p][c] = (sum_k e_k * vp_k) / (sum_k e_k),  e_k = exp(s_k - max)
    for (int t = tid; t < NB * DIM; t += 256) {
        const int p = t >> 8;
        const int c = t & (DIM - 1);
        const float* col = SC + (size_t)(p * KNN) * DIM + c;
        float m = -3.402823e38f;
        for (int kk = 0; kk < KNN; ++kk)
            m = fmaxf(m, col[kk * DIM]);
        float s = 0.0f, acc = 0.0f;
        for (int kk = 0; kk < KNN; ++kk) {
            float e = __expf(col[kk * DIM] - m);
            s += e;
            acc += e * bf2f(VP[(p * KNN + kk) * DIM + c]);
        }
        resb[(size_t)(n0 + p) * DIM + c] = f2bf(acc / s);
    }
}

// ---------------------------------------------------------------------------
// Host launcher
// ---------------------------------------------------------------------------
extern "C" void kernel_launch(void* const* d_in, const int* in_sizes, int n_in,
                              void* d_out, int out_size, void* d_ws, size_t ws_size,
                              hipStream_t stream) {
    const float* xyz_q    = (const float*)d_in[0];
    const float* feats_q  = (const float*)d_in[1];
    const float* xyz_kv   = (const float*)d_in[2];
    const float* feats_kv = (const float*)d_in[3];
    const float* Wq   = (const float*)d_in[4];
    const float* Wk   = (const float*)d_in[5];
    const float* Wv   = (const float*)d_in[6];
    const float* d1_w = (const float*)d_in[7];
    const float* d1_b = (const float*)d_in[8];
    const float* d2_w = (const float*)d_in[9];
    const float* d2_b = (const float*)d_in[10];
    const float* g1_w = (const float*)d_in[11];
    const float* g1_b = (const float*)d_in[12];
    const float* g2_w = (const float*)d_in[13];
    const float* g2_b = (const float*)d_in[14];
    const float* out_w = (const float*)d_in[15];
    const float* out_b = (const float*)d_in[16];

    char* ws = (char*)d_ws;
    // workspace layout (all offsets 256B aligned)
    __bf16* WQT = (__bf16*)(ws + 0);                      // 256x256
    __bf16* WKT = (__bf16*)(ws + 131072);                 // 256x256
    __bf16* WVT = (__bf16*)(ws + 262144);                 // 256x256
    __bf16* D2T = (__bf16*)(ws + 393216);                 // 256x128
    __bf16* G1T = (__bf16*)(ws + 458752);                 // 256x256
    __bf16* G2T = (__bf16*)(ws + 589824);                 // 256x256
    __bf16* OWT = (__bf16*)(ws + 720896);                 // 512x256
    __bf16* FQB = (__bf16*)(ws + 983040);                 // 8192x256 bf16
    float*  QAT = (float*) (ws + 5177344);                // 8192x256 f32
    __bf16* RSB = (__bf16*)(ws + 13565952);               // 8192x256 bf16

    float* out_xyz = (float*)d_out;                       // 8192*3
    float* out_res = (float*)d_out + NPTS * 3;            // 8192*512

    // prep: weight transpose + convert; feats_q convert; xyz passthrough
    k_transpose_bf16<<<256, 256, 0, stream>>>(Wq,    WQT, DIM,  DIM);
    k_transpose_bf16<<<256, 256, 0, stream>>>(Wk,    WKT, DIM,  DIM);
    k_transpose_bf16<<<256, 256, 0, stream>>>(Wv,    WVT, DIM,  DIM);
    k_transpose_bf16<<<128, 256, 0, stream>>>(d2_w,  D2T, DIMH, DIM);
    k_transpose_bf16<<<256, 256, 0, stream>>>(g1_w,  G1T, DIM,  DIM);
    k_transpose_bf16<<<256, 256, 0, stream>>>(g2_w,  G2T, DIM,  DIM);
    k_transpose_bf16<<<512, 256, 0, stream>>>(out_w, OWT, DIM,  DIMOUT);
    k_conv_bf16<<<2048, 256, 0, stream>>>(feats_q, FQB, NPTS * DIM);
    k_copy_f32<<<96, 256, 0, stream>>>(xyz_q, out_xyz, NPTS * 3);

    // q_attn = feats_q @ Wq  (no bias)
    gemm_bf16<<<NPTS / 64, 256, 0, stream>>>(FQB, WQT, nullptr, QAT,
                                             DIM, DIM);

    // fused attention (176 KB dynamic LDS of the 320 KB WGP pool)
    (void)hipFuncSetAttribute((const void*)fused_attn,
                              hipFuncAttributeMaxDynamicSharedMemorySize,
                              LDS_TOT);
    fused_attn<<<NPTS / NB, 256, LDS_TOT, stream>>>(
        xyz_q, xyz_kv, feats_kv, d1_w, d1_b, d2_b, g1_b, g2_b,
        WKT, WVT, D2T, G1T, G2T, QAT, RSB);

    // out = res @ out_w + out_b
    gemm_bf16<<<NPTS / 64, 256, 0, stream>>>(RSB, OWT, out_b, out_res,
                                             DIMOUT, DIMOUT);
}
